// SNNModel_18906446037610
// MI455X (gfx1250) — matrix operand order repaired and checked by reading, not compiled, not run on hardware
//
#include <hip/hip_runtime.h>

typedef __attribute__((ext_vector_type(16))) _Float16     v16h;
typedef __attribute__((ext_vector_type(8)))  float        v8f;
typedef __attribute__((ext_vector_type(4)))  unsigned int u32x4;
typedef __attribute__((ext_vector_type(8)))  unsigned int u32x8;

#define BATCH  512
#define TSTEPS 1000
#define DIN    20
#define DH     600
#define DOUT   100
#define DH_P   640      // layer-1 padded width: 40 tiles of 16
#define K2     608      // layer-2 K extent (>= 600), 19 chunks of 32
#define NCH2   (K2/32)  // 19
#define NT2    7        // layer-2 N tiles (112 >= 100)
#define BETA_F 0.95f
#define TH1_F  0.3f
#define TH2_F  0.5f

static __device__ __forceinline__ void wait_async0() {
#if __has_builtin(__builtin_amdgcn_s_wait_asynccnt)
    __builtin_amdgcn_s_wait_asynccnt(0);
#else
    asm volatile("s_wait_asynccnt 0x0" ::: "memory");
#endif
}

// TDM: DMA one 16x20 f32 tile (row stride 20000 f32) from global into LDS,
// zero-padded by hardware OOB handling to a packed 16x32 f32 block.
// D# group0: count=1 | lds_addr | global_addr | type=2
// D# group1: data_size=4B, tensor_dim0=20, tensor_dim1=16, tile_dim0=32,
//            tile_dim1=16, tensor_dim0_stride=20000 (2D -> groups 2/3 NULL)
static __device__ __forceinline__ void tdm_load_x(const float* gptr, unsigned lds_off) {
    unsigned long long ga = (unsigned long long)gptr;
    u32x4 g0;
    g0[0] = 0x1u;                                   // count=1, user mode
    g0[1] = lds_off;                                // LDS byte address
    g0[2] = (unsigned)ga;                           // global_addr[31:0]
    g0[3] = (unsigned)(ga >> 32) | 0x80000000u;     // global_addr[56:32] | type=2
    u32x8 g1;
    g1[0] = 0x00020000u;                            // data_size=2 (4 bytes)
    g1[1] = (unsigned)DIN << 16;                    // tensor_dim0[15:0] = 20
    g1[2] = 16u << 16;                              // tensor_dim1[15:0] = 16
    g1[3] = 32u << 16;                              // tile_dim0 = 32 (cols 20..31 -> OOB zeros)
    g1[4] = 16u;                                    // tile_dim1 = 16
    g1[5] = (unsigned)(TSTEPS * DIN);               // tensor_dim0_stride = 20000
    g1[6] = 0u;
    g1[7] = 0u;
    asm volatile("tensor_load_to_lds %0, %1" :: "s"(g0), "s"(g1) : "memory");
}

static __device__ __forceinline__ void wait_tensor0() {
#if __has_builtin(__builtin_amdgcn_s_wait_tensorcnt)
    __builtin_amdgcn_s_wait_tensorcnt(0);
#else
    asm volatile("s_wait_tensorcnt 0x0" ::: "memory");
#endif
}

// One workgroup = 16 batch rows, 8 wave32's, runs all T=1000 steps privately.
// State in VGPRs; W1 fragments in VGPRs; W2 (f16) in LDS; x tile double-buffered
// in LDS via per-step TDM DMA; biases staged once via async global->LDS loads.
__launch_bounds__(256, 1)
__global__ void snn_persistent_kernel(const float* __restrict__ x,
                                      const float* __restrict__ W1,
                                      const float* __restrict__ b1,
                                      const float* __restrict__ W2,
                                      const float* __restrict__ b2,
                                      float* __restrict__ out)
{
    __shared__ _Float16 sW2[NT2 * 16][K2];    // [n][k], 136,192 B
    __shared__ _Float16 sSpk[16][DH_P];       // layer-2 A source,  20,480 B
    __shared__ float    sXf32[2][16][32];     // TDM x staging,      4,096 B
    __shared__ float    sBias[DH + DOUT];     // async bias staging, 2,800 B

    const int tid  = threadIdx.x;
    const int wave = tid >> 5;
    const int lane = tid & 31;
    const int ln   = lane & 15;   // N (B/C/D) or M (A) index within a tile
    const int hf   = lane >> 4;   // lane half
    const int b0   = blockIdx.x << 4;

    // ---- stage biases via async global->LDS loads (ASYNCcnt-tracked)
    for (int i = tid; i < DH + DOUT; i += 256) {
        unsigned lds = (unsigned)(unsigned long long)&sBias[i];
        const float* g = (i < DH) ? (b1 + i) : (b2 + (i - DH));
        asm volatile("global_load_async_to_lds_b32 %0, %1, off"
                     :: "v"(lds), "v"((unsigned long long)g) : "memory");
    }

    // ---- zero x staging buffers (TDM re-zeros pads each step; belt & braces)
    for (int i = tid; i < 2 * 16 * 32; i += 256)
        ((float*)sXf32)[i] = 0.f;

    // ---- stage W2 (f32 -> f16) into LDS as [n][k] (B = W2^T, so B[k][n] = W2[n][k])
    for (int n = wave; n < NT2 * 16; n += 8)
        for (int k = lane; k < K2; k += 32) {
            float w = (n < DOUT && k < DH) ? W2[n * DH + k] : 0.f;
            sW2[n][k] = (_Float16)w;
        }

    // ---- per-wave W1 fragments (f16 B-matrix 32x16 layout), VGPR-resident all T steps
    v16h w1f[5];
#pragma unroll
    for (int j = 0; j < 5; ++j) {
        int n1 = wave * 80 + j * 16 + ln;          // neuron (N) index, padded to 640
#pragma unroll
        for (int p = 0; p < 8; ++p) {
            // B 32x16: lanes 0-15 hold K=0..15 (2 per VGPR), lanes 16-31 K=16..31
            int k0 = (p << 1) + (hf << 4);
            _Float16 e0 = (_Float16)0.f, e1 = (_Float16)0.f;
            if (n1 < DH) {
                if (k0     < DIN) e0 = (_Float16)W1[n1 * DIN + k0];
                if (k0 + 1 < DIN) e1 = (_Float16)W1[n1 * DIN + k0 + 1];
            }
            w1f[j][2 * p]     = e0;
            w1f[j][2 * p + 1] = e1;
        }
    }

    wait_async0();                  // this wave's bias loads landed in LDS
    __syncthreads();                // everything staged & visible

    // ---- biases from LDS into registers
    float bias1[5];
#pragma unroll
    for (int j = 0; j < 5; ++j) {
        int n1 = wave * 80 + j * 16 + ln;
        bias1[j] = (n1 < DH) ? sBias[n1] : 0.f;
    }
    const int   n2    = wave * 16 + ln;            // layer-2 output column
    const float bias2 = (wave < NT2 && n2 < DOUT) ? sBias[DH + n2] : 0.f;

    // ---- recurrent state in VGPRs
    v8f mem1[5], spk1[5], mem2, spk2;
#pragma unroll
    for (int j = 0; j < 5; ++j)
#pragma unroll
        for (int r = 0; r < 8; ++r) { mem1[j][r] = 0.f; spk1[j][r] = 0.f; }
#pragma unroll
    for (int r = 0; r < 8; ++r) { mem2[r] = 0.f; spk2[r] = 0.f; }

    // ---- kick TDM for t = 0 into buffer 0 (wave 0 drives the data mover)
    if (wave == 0)
        tdm_load_x(x + (long)b0 * (TSTEPS * DIN),
                   (unsigned)(unsigned long long)&sXf32[0][0][0]);

    for (int t = 0; t < TSTEPS; ++t) {
        wait_tensor0();             // step-t tile DMA complete (non-issuers pass free)
        __syncthreads();            // tile visible to all waves; fences sSpk/sXf32 reuse

        // DMA x[:, t+1, :] into the other buffer; overlaps this step's compute
        if (wave == 0 && t + 1 < TSTEPS)
            tdm_load_x(x + (long)b0 * (TSTEPS * DIN) + (long)(t + 1) * DIN,
                       (unsigned)(unsigned long long)&sXf32[(t + 1) & 1][0][0]);
        if (t + 4 < TSTEPS)         // warm L2 ahead of the TDM pipeline
            __builtin_prefetch(&x[(long)(b0 + ln) * (TSTEPS * DIN) + (long)(t + 4) * DIN + hf * 16], 0, 0);

        // ================= layer 1: cur1 = x_t @ W1^T + b1 =================
        // A 16x32 f16 fragment (contiguous layout: VGPR p -> K = p*2 + hf*16)
        const float* xr = &sXf32[t & 1][ln][0];
        v16h ax;
#pragma unroll
        for (int p = 0; p < 8; ++p) {
            int k0 = (p << 1) + (hf << 4);
            ax[2 * p]     = (_Float16)xr[k0];
            ax[2 * p + 1] = (_Float16)xr[k0 + 1];
        }
#pragma unroll
        for (int j = 0; j < 5; ++j) {
            v8f acc;
#pragma unroll
            for (int r = 0; r < 8; ++r) acc[r] = bias1[j];
            acc = __builtin_amdgcn_wmma_f32_16x16x32_f16(false, ax, false, w1f[j],
                                                         (short)0, acc, false, false);
            int nb = wave * 80 + j * 16 + ln;
#pragma unroll
            for (int r = 0; r < 8; ++r) {
                float nm   = BETA_F * mem1[j][r] + acc[r] - spk1[j][r] * TH1_F;
                mem1[j][r] = nm;
                float s    = (nm > TH1_F) ? 1.f : 0.f;   // spike exact in f16
                spk1[j][r] = s;
                sSpk[r + 8 * hf][nb] = (_Float16)s;      // C layout: M = r + 8*hf, N = nb
            }
        }
        __syncthreads();

        // ================= layer 2: cur2 = spk1 @ W2^T + b2 =================
        if (wave < NT2) {
            v8f acc0, acc1;
#pragma unroll
            for (int r = 0; r < 8; ++r) { acc0[r] = bias2; acc1[r] = 0.f; }
#pragma unroll
            for (int c = 0; c < NCH2; ++c) {
                const _Float16* ar = &sSpk[ln][c * 32];
                const _Float16* br = &sW2[n2][c * 32];
                v16h a, b;
#pragma unroll
                for (int p = 0; p < 8; ++p) {
                    int ka = (p << 1) + (hf << 4);
                    a[2 * p]     = ar[ka];
                    a[2 * p + 1] = ar[ka + 1];
                    int kb = (p << 1) + (hf << 4);
                    b[2 * p]     = br[kb];
                    b[2 * p + 1] = br[kb + 1];
                }
                if (c & 1)
                    acc1 = __builtin_amdgcn_wmma_f32_16x16x32_f16(false, a, false, b,
                                                                  (short)0, acc1, false, false);
                else
                    acc0 = __builtin_amdgcn_wmma_f32_16x16x32_f16(false, a, false, b,
                                                                  (short)0, acc0, false, false);
            }
#pragma unroll
            for (int r = 0; r < 8; ++r) {
                float cur = acc0[r] + acc1[r];
                float nm  = BETA_F * mem2[r] + cur - spk2[r] * TH2_F;
                mem2[r]   = nm;
                spk2[r]   = (nm > TH2_F) ? 1.f : 0.f;
            }
            if (n2 < DOUT) {
#pragma unroll
                for (int r = 0; r < 8; ++r) {
                    int m = r + 8 * hf;
                    out[(long)t * (BATCH * DOUT) + (long)(b0 + m) * DOUT + n2] = mem2[r];
                }
            }
        }
        // no trailing barrier: next iteration's wait+barrier fences sSpk / sXf32 reuse
    }
}

extern "C" void kernel_launch(void* const* d_in, const int* in_sizes, int n_in,
                              void* d_out, int out_size, void* d_ws, size_t ws_size,
                              hipStream_t stream) {
    const float* x  = (const float*)d_in[0];
    const float* W1 = (const float*)d_in[1];
    const float* b1 = (const float*)d_in[2];
    const float* W2 = (const float*)d_in[3];
    const float* b2 = (const float*)d_in[4];
    float* out = (float*)d_out;
    snn_persistent_kernel<<<BATCH / 16, 256, 0, stream>>>(x, W1, b1, W2, b2, out);
}
